// LinearAttention_46042049413570
// MI455X (gfx1250) — matrix operand order repaired and checked by reading, not compiled, not run on hardware
//
#include <hip/hip_runtime.h>

#define BZ   1024
#define DNUM 200
#define FEAT 256
#define HID  256
#define KTOT 512            // FEAT + HID concatenated K
#define LDK  (KTOT + 2)     // padded K stride (bf16 elems) -> bank-conflict-free
#define MT   13             // ceil(DNUM/16)

typedef __attribute__((ext_vector_type(8)))  float  v8f;
typedef __attribute__((ext_vector_type(16))) __bf16 v16bf;

union BfVec { v16bf v; unsigned int u[8]; };

__device__ __forceinline__ unsigned short f2bf(float f) {
  unsigned int x = __float_as_uint(f);
  unsigned int r = x + 0x7FFFu + ((x >> 16) & 1u);   // round-to-nearest-even
  return (unsigned short)(r >> 16);
}

#define BS_ELEMS (HID * LDK)    // weights in LDS, K-contiguous per output column
#define AS_ELEMS (16 * LDK)     // one 16-row A tile
#define SMEM_BYTES (BS_ELEMS * 2 + AS_ELEMS * 2 + 208 * 4 + 208 * 4 + 256 * 4)

__global__ void __launch_bounds__(256)
linear_attention_kernel(const float* __restrict__ feat,
                        const int*   __restrict__ other_id,
                        const float* __restrict__ W_rv,
                        const float* __restrict__ W_id,
                        const float* __restrict__ hvec,
                        const float* __restrict__ b1,
                        const float* __restrict__ b2,
                        const float* __restrict__ ebd,
                        float* __restrict__ out,       // [BZ][FEAT]
                        float* __restrict__ out_sc)    // [BZ][DNUM]
{
  extern __shared__ char smem[];
  unsigned short* Bs = (unsigned short*)smem;          // [HID][LDK] bf16
  unsigned short* As = Bs + BS_ELEMS;                  // [16][LDK]  bf16
  float* logits = (float*)(As + AS_ELEMS);             // [208]
  float* scores = logits + 208;                        // [208]
  float* red    = scores + 208;                        // [256]

  const int b    = blockIdx.x;
  const int tid  = threadIdx.x;
  const int wave = tid >> 5;
  const int lane = tid & 31;
  const int m    = lane & 15;     // M for A-frag / N for B-frag / N for C
  const int hi   = lane >> 4;     // K half selector

  // ---- Phase 0: stage [W_rv ; W_id] fp32 -> bf16 into LDS (coalesced reads) ----
  for (int e = tid; e < KTOT * HID; e += 256) {
    int k = e >> 8;               // 0..511
    int n = e & 255;              // output column
    float w = (k < FEAT) ? W_rv[k * HID + n] : W_id[(k - FEAT) * HID + n];
    Bs[n * LDK + k] = f2bf(w);
  }
  if (tid < 208) logits[tid] = b2[0];
  __syncthreads();

  // ---- Phase 1: 13 M-tiles; WMMA GEMM fused with bias+relu+dot(h) ----
  for (int t = 0; t < MT; ++t) {
    const int d0 = t * 16;
    // cooperative 16x512 A tile: [feat | gathered embedding], zero-padded
    for (int e = tid; e < 16 * KTOT; e += 256) {
      int row = e >> 9;
      int col = e & 511;
      int d = d0 + row;
      float v = 0.f;
      if (d < DNUM) {
        if (col < FEAT) {
          v = feat[((size_t)(b * DNUM + d)) * FEAT + col];
        } else {
          int id = other_id[b * DNUM + d];
          if (id != 0) v = ebd[(size_t)id * HID + (col - FEAT)];  // padding_idx=0
        }
      }
      As[row * LDK + col] = f2bf(v);
    }
    __syncthreads();

    for (int nt = wave; nt < 16; nt += 8) {       // 2 N-tiles per wave
      const int nbase = nt * 16;
      v8f acc = {};
      for (int kb = 0; kb < 16; ++kb) {           // K = 512 in steps of 32
        const int kbase = kb * 32;
        BfVec a, bv;
        #pragma unroll
        for (int r = 0; r < 8; ++r) {
          // A 16x32 bf16, contiguous k order used here (same as B): lanes 0-15 K=0..15, lanes 16-31 K=16..31
          int ka = kbase + 16 * hi + 2 * r;
          a.u[r] = *(const unsigned int*)&As[m * LDK + ka];
          // B 32x16 bf16, contiguous k order used here (same as A): lanes 0-15 K=0..15, lanes 16-31 K=16..31
          int kn = kbase + 16 * hi + 2 * r;
          bv.u[r] = *(const unsigned int*)&Bs[(nbase + m) * LDK + kn];
        }
        acc = __builtin_amdgcn_wmma_f32_16x16x32_bf16(
            false, a.v, false, bv.v, (short)0, acc, false, false);
      }
      // epilogue: C lane holds rows M = hi*8 + v, col N = nbase + m
      const int nn = nbase + m;
      const float bias = b1[nn];
      const float hv   = hvec[nn];
      #pragma unroll
      for (int v = 0; v < 8; ++v) {
        float p = acc[v] + bias;
        p = p > 0.f ? p : 0.f;
        float s = p * hv;
        s += __shfl_xor(s, 1, 32);                // reduce over 16 N-lanes
        s += __shfl_xor(s, 2, 32);
        s += __shfl_xor(s, 4, 32);
        s += __shfl_xor(s, 8, 32);
        if (m == 0) atomicAdd(&logits[d0 + hi * 8 + v], s);
      }
    }
    __syncthreads();
  }

  // ---- Phase 2: e = exp(logit); scores = e / (sum e + 1e-8) (no max-sub) ----
  if (tid < DNUM) scores[tid] = expf(logits[tid]);
  red[tid] = (tid < DNUM) ? scores[tid] : 0.f;
  __syncthreads();
  for (int s = 128; s > 0; s >>= 1) {
    if (tid < s) red[tid] += red[tid + s];
    __syncthreads();
  }
  const float denom = red[0] + 1e-8f;
  if (tid < DNUM) {
    float sc = scores[tid] / denom;
    scores[tid] = sc;
    out_sc[(size_t)b * DNUM + tid] = sc;
  }
  __syncthreads();

  // ---- Phase 3: out[b][f] = sum_d score[d] * feat[b][d][f] (coalesced, L2-hot) ----
  const float* fb = feat + (size_t)b * DNUM * FEAT + tid;
  float accv = 0.f;
  #pragma unroll 4
  for (int d = 0; d < DNUM; ++d) {
    accv += scores[d] * fb[(size_t)d * FEAT];
  }
  out[(size_t)b * FEAT + tid] = accv;
}

extern "C" void kernel_launch(void* const* d_in, const int* in_sizes, int n_in,
                              void* d_out, int out_size, void* d_ws, size_t ws_size,
                              hipStream_t stream) {
  const float* feat     = (const float*)d_in[0];
  const int*   other_id = (const int*)d_in[1];
  const float* W_rv     = (const float*)d_in[2];
  const float* W_id     = (const float*)d_in[3];
  const float* hvec     = (const float*)d_in[4];
  const float* b1       = (const float*)d_in[5];
  const float* b2       = (const float*)d_in[6];
  const float* ebd      = (const float*)d_in[7];
  float* out    = (float*)d_out;                  // [BZ*FEAT] first
  float* out_sc = out + (size_t)BZ * FEAT;        // then [BZ*DNUM] scores

  dim3 grid(BZ), block(256);
  linear_attention_kernel<<<grid, block, SMEM_BYTES, stream>>>(
      feat, other_id, W_rv, W_id, hvec, b1, b2, ebd, out, out_sc);
}
